// SAE_33466385170567
// MI455X (gfx1250) — matrix operand order repaired and checked by reading, not compiled, not run on hardware
//
#include <hip/hip_runtime.h>
#include <hip/hip_bf16.h>
#include <stdint.h>

#define D_MODEL 2048
#define D_SAE   16384
#define N_TOK   8192
#define TOPK    128

typedef __attribute__((ext_vector_type(16))) __bf16 v16bf;
typedef __attribute__((ext_vector_type(8)))  float  v8f;
typedef __attribute__((ext_vector_type(4)))  unsigned v4u;
typedef __attribute__((ext_vector_type(8)))  unsigned v8u;

union Frag16 { v16bf v; unsigned u[8]; };

__device__ __forceinline__ unsigned pack2_bf16(float a, float b) {
    unsigned ua = __float_as_uint(a);
    unsigned ub = __float_as_uint(b);
    ua = (ua + 0x7FFFu + ((ua >> 16) & 1u)) >> 16;   // round-to-nearest-even
    ub = (ub + 0x7FFFu + ((ub >> 16) & 1u)) >> 16;
    return (ua & 0xFFFFu) | (ub << 16);
}

// 32-bit LDS offset of a generic pointer into shared memory.
__device__ __forceinline__ unsigned lds_off(const void* p) {
    return (unsigned)(size_t)(__attribute__((address_space(3))) const void*)p;
}

// Per-lane async DMA: 16 bytes global -> LDS, tracked by ASYNCcnt.
__device__ __forceinline__ void async_b128(unsigned lds, const void* g) {
    asm volatile("global_load_async_to_lds_b128 %0, %1, off"
                 :: "v"(lds), "v"(g) : "memory");
}

// ---------------------------------------------------------------------------
// f32 -> bf16 bulk convert (8 elems/thread); n must be a multiple of 8.
// ---------------------------------------------------------------------------
__global__ __launch_bounds__(256)
void k_convert(const float* __restrict__ src, unsigned short* __restrict__ dst) {
    const size_t i = ((size_t)blockIdx.x * 256 + threadIdx.x) * 8;
    const float4 a = *(const float4*)(src + i);
    const float4 b = *(const float4*)(src + i + 4);
    uint4 o;
    o.x = pack2_bf16(a.x, a.y); o.y = pack2_bf16(a.z, a.w);
    o.z = pack2_bf16(b.x, b.y); o.w = pack2_bf16(b.z, b.w);
    *(uint4*)(dst + i) = o;
}

// ---------------------------------------------------------------------------
// b_eff[j] = b_enc[j] - sum_d b_dec[d] * W_enc[j][d]      (one wave32 per j)
// ---------------------------------------------------------------------------
__global__ __launch_bounds__(256)
void k_beff(const float* __restrict__ W_enc,
            const float* __restrict__ b_enc,
            const float* __restrict__ b_dec,
            float* __restrict__ beff) {
    const int wave = threadIdx.x >> 5;
    const int lane = threadIdx.x & 31;
    const int j = blockIdx.x * 8 + wave;
    const float* wr = W_enc + (size_t)j * D_MODEL;
    float s = 0.f;
    for (int d = lane; d < D_MODEL; d += 32) s += wr[d] * b_dec[d];
    for (int off = 16; off > 0; off >>= 1) s += __shfl_xor(s, off, 32);
    if (lane == 0) beff[j] = b_enc[j] - s;
}

// ---------------------------------------------------------------------------
// W_decT[j][d] = W_dec[d][j]   (32x32 LDS tiles; W_decT = 128 MB, L2-resident)
// ---------------------------------------------------------------------------
__global__ __launch_bounds__(256)
void k_transpose(const float* __restrict__ W_dec, float* __restrict__ WdT) {
    __shared__ float t[32][33];
    const int jx = blockIdx.x * 32, dy = blockIdx.y * 32;
    const int tx = threadIdx.x, ty = threadIdx.y;
#pragma unroll
    for (int r = 0; r < 4; ++r)
        t[ty + 8 * r][tx] = W_dec[(size_t)(dy + ty + 8 * r) * D_SAE + jx + tx];
    __syncthreads();
#pragma unroll
    for (int r = 0; r < 4; ++r)
        WdT[(size_t)(jx + ty + 8 * r) * D_MODEL + dy + tx] = t[tx][ty + 8 * r];
}

// ---------------------------------------------------------------------------
// Encoder GEMM: enc = relu(x * W_enc^T + b_eff), bf16 WMMA, f32 accumulate.
// Double-buffered GLOBAL_LOAD_ASYNC_TO_LDS_B128 staging (ASYNCcnt); last
// iteration peeled so the steady-state loop body is branch-free.
// Block tile 128x128, 8 waves; each wave 64x32 = 4x2 v_wmma_f32_16x16x32_bf16.
// ---------------------------------------------------------------------------
#define BM 128
#define BN 128
#define BK 32
#define LDA 40  // padded LDS row stride (bf16 elems): 80B rows, 16B aligned

__global__ __launch_bounds__(256)
void k_encoder(const unsigned short* __restrict__ xb,
               const unsigned short* __restrict__ wb,
               const float* __restrict__ beff,
               float* __restrict__ enc) {
    __shared__ unsigned short As[2][BM * LDA];
    __shared__ unsigned short Bs[2][BN * LDA];

    const int tid  = threadIdx.x;
    const int lane = tid & 31;
    const int wave = tid >> 5;
    const int wm   = wave & 1;    // 2 waves along M (64 rows each)
    const int wn   = wave >> 1;   // 4 waves along N (32 cols each)
    const int half = lane >> 4;
    const int l15  = lane & 15;

    const int gm0 = blockIdx.y * BM;
    const int gn0 = blockIdx.x * BN;

    v8f acc[4][2] = {};

    // staging map: 128 rows x 32 bf16 (64B) per tile; 16B chunk per lane-slot
    auto issue = [&](int k0, int buf) {
#pragma unroll
        for (int r = 0; r < 2; ++r) {
            const int slot = tid + 256 * r;
            const int row  = slot >> 2;
            const int c8   = (slot & 3) * 8;
            async_b128(lds_off(&As[buf][row * LDA + c8]),
                       xb + (size_t)(gm0 + row) * D_MODEL + k0 + c8);
            async_b128(lds_off(&Bs[buf][row * LDA + c8]),
                       wb + (size_t)(gn0 + row) * D_MODEL + k0 + c8);
        }
    };

    auto compute = [&](int buf) {
        // B fragments: lane holds col l15; uint slot v <- K pair (2v + 16*half)
        Frag16 bf[2];
#pragma unroll
        for (int tn = 0; tn < 2; ++tn) {
            const unsigned* bp = (const unsigned*)&Bs[buf][(wn * 32 + tn * 16 + l15) * LDA];
#pragma unroll
            for (int v = 0; v < 8; ++v) bf[tn].u[v] = bp[v + 8 * half];
        }
        // A fragments: lane holds row l15; uint slot v <- K pair (2v + 16*half)
#pragma unroll
        for (int tm = 0; tm < 4; ++tm) {
            Frag16 af;
            const unsigned* ap = (const unsigned*)&As[buf][(wm * 64 + tm * 16 + l15) * LDA];
#pragma unroll
            for (int v = 0; v < 8; ++v) af.u[v] = ap[v + 8 * half];
#pragma unroll
            for (int tn = 0; tn < 2; ++tn) {
                acc[tm][tn] = __builtin_amdgcn_wmma_f32_16x16x32_bf16(
                    false, af.v, false, bf[tn].v, (short)0, acc[tm][tn], false, false);
            }
        }
    };

    issue(0, 0);
    int buf = 0;
    for (int k0 = 0; k0 < D_MODEL - BK; k0 += BK) {   // steady state: branch-free
        issue(k0 + BK, buf ^ 1);
        asm volatile("s_wait_asynccnt 0x4" ::: "memory");  // previous batch done
        __syncthreads();
        compute(buf);
        __syncthreads();   // all waves done reading before next overwrite
        buf ^= 1;
    }
    asm volatile("s_wait_asynccnt 0x0" ::: "memory");       // peeled last tile
    __syncthreads();
    compute(buf);

    // epilogue: + b_eff, ReLU, store dense f32
#pragma unroll
    for (int tm = 0; tm < 4; ++tm) {
#pragma unroll
        for (int tn = 0; tn < 2; ++tn) {
            const int col = gn0 + wn * 32 + tn * 16 + l15;
            const float be = beff[col];
#pragma unroll
            for (int v = 0; v < 8; ++v) {
                const int row = gm0 + wm * 64 + tm * 16 + v + 8 * half;
                enc[(size_t)row * D_SAE + col] = fmaxf(acc[tm][tn][v] + be, 0.0f);
            }
        }
    }
}

// ---------------------------------------------------------------------------
// Exact top-k per token via 3-pass radix select on float bits (values >= 0
// after ReLU). The 64 KB row is moved to LDS by the Tensor Data Mover:
// one tensor_load_to_lds per workgroup (D# per CDNA5 ISA
// section 8; 1-row tile of 16384 x 4B), waited with s_wait_tensorcnt.
// Rewrites the dense row sparsified in place + emits compact (idx,val) lists.
// ---------------------------------------------------------------------------
__global__ __launch_bounds__(256)
void k_topk(float* __restrict__ enc,
            float* __restrict__ topv, int* __restrict__ topi) {
    extern __shared__ __align__(16) char smem[];
    float* row  = (float*)smem;                 // 16384 f32 (64 KB)
    int*   hist = (int*)(smem + D_SAE * 4);     // 2048 bins
    int*   sidx = hist + 2048;                  // 128
    float* sval = (float*)(sidx + TOPK);        // 128
    __shared__ int s_bin, s_above, c_gt, c_eq;

    const int tid = threadIdx.x;
    float* g = enc + (size_t)blockIdx.x * D_SAE;

    if (tid < 32) {  // wave 0 drives the TDM (EXEC ignored; one issue per WG)
        const unsigned long long ga = (unsigned long long)(size_t)g;
        const unsigned la = lds_off(row);
        // D# group0: count=1 | lds_addr | global_addr[56:0] | type=2
        const v4u g0 = { 1u, la, (unsigned)ga,
                         (unsigned)((ga >> 32) & 0x1FFFFFFu) | 0x80000000u };
        // D# group1: data_size=4B; tensor_dim0=16384, dim1=1; tile_dim0=16384,
        // tile_dim1=1, tile_dim2=0; dim0_stride=16384; dim1_stride=0
        const v8u g1 = { 0x00020000u, 0x40000000u, 0x00010000u, 0x40000000u,
                         0x00000001u, 0x00004000u, 0u, 0u };
        // D# group2: tensor_dim2=1, tensor_dim3=1, dim2_stride=0, tile_dim3=0
        const v4u g2 = { 1u, 1u, 0u, 0u };
        // D# group3: dim3_stride=0, tensor_dim4=1, tile_dim4=0
        const v4u g3 = { 0u, 0x00010000u, 0u, 0u };
        asm volatile("tensor_load_to_lds %0, %1, %2, %3"
                     :: "s"(g0), "s"(g1), "s"(g2), "s"(g3) : "memory");
        __builtin_amdgcn_s_wait_tensorcnt(0);
    }
    __syncthreads();

    unsigned prefix = 0, pmask = 0;
    int need = TOPK;
    const int shifts[3] = {21, 10, 0};
    const int nbins[3]  = {2048, 2048, 1024};

    for (int pass = 0; pass < 3; ++pass) {
        const int s = shifts[pass], nb = nbins[pass];
        for (int i = tid; i < nb; i += 256) hist[i] = 0;
        __syncthreads();
        for (int i = tid; i < D_SAE; i += 256) {
            const unsigned key = __float_as_uint(row[i]);
            if ((key & pmask) == prefix)
                atomicAdd(&hist[(key >> s) & (nb - 1)], 1);
        }
        __syncthreads();
        if (tid == 0) {
            int cum = 0, b = nb - 1;
            for (; b > 0; --b) {
                const int c = hist[b];
                if (cum + c >= need) break;
                cum += c;
            }
            s_bin = b; s_above = cum;
        }
        __syncthreads();
        prefix |= ((unsigned)s_bin) << s;
        pmask  |= ((unsigned)(nb - 1)) << s;
        need   -= s_above;
        __syncthreads();
    }

    const unsigned T = prefix;        // exact k-th largest key
    const int G = TOPK - need;        // count strictly greater than T
    if (tid == 0) { c_gt = 0; c_eq = 0; }
    __syncthreads();
    for (int i = tid; i < D_SAE; i += 256) {
        const unsigned key = __float_as_uint(row[i]);
        if (key > T) {
            const int p = atomicAdd(&c_gt, 1);
            sidx[p] = i; sval[p] = row[i];
        } else if (key == T) {
            const int e = atomicAdd(&c_eq, 1);
            if (e < need) { sidx[G + e] = i; sval[G + e] = row[i]; }
        }
    }
    for (int i = tid; i < D_SAE; i += 256) g[i] = 0.0f;  // sparsify row
    __syncthreads();
    if (tid < TOPK) {
        const int j = sidx[tid];
        const float v = sval[tid];
        g[j] = v;
        topv[blockIdx.x * TOPK + tid] = v;
        topi[blockIdx.x * TOPK + tid] = j;
    }
}

// ---------------------------------------------------------------------------
// Sparse decode: out[tok] = b_dec + sum_i val_i * W_decT[idx_i]  (L2-resident)
// 256 threads per token; thread t owns d = t + 256*u, fully coalesced rows.
// ---------------------------------------------------------------------------
__global__ __launch_bounds__(256)
void k_decode(const float* __restrict__ WdT,
              const float* __restrict__ b_dec,
              const float* __restrict__ topv, const int* __restrict__ topi,
              float* __restrict__ out) {
    __shared__ float sv[TOPK];
    __shared__ int   si[TOPK];
    const int tid = threadIdx.x;
    const int tok = blockIdx.x;
    if (tid < TOPK) {
        sv[tid] = topv[tok * TOPK + tid];
        si[tid] = topi[tok * TOPK + tid];
    }
    __syncthreads();
    float acc[8];
#pragma unroll
    for (int u = 0; u < 8; ++u) acc[u] = b_dec[tid + 256 * u];
    for (int i = 0; i < TOPK; ++i) {
        const float v = sv[i];
        const float* wr = WdT + (size_t)si[i] * D_MODEL;
        if (i + 1 < TOPK)
            __builtin_prefetch(WdT + (size_t)si[i + 1] * D_MODEL + tid, 0, 0);
#pragma unroll
        for (int u = 0; u < 8; ++u) acc[u] += v * wr[tid + 256 * u];
    }
#pragma unroll
    for (int u = 0; u < 8; ++u)
        out[(size_t)tok * D_MODEL + tid + 256 * u] = acc[u];
}

// ---------------------------------------------------------------------------
extern "C" void kernel_launch(void* const* d_in, const int* in_sizes, int n_in,
                              void* d_out, int out_size, void* d_ws, size_t ws_size,
                              hipStream_t stream) {
    (void)in_sizes; (void)n_in; (void)out_size; (void)ws_size;
    const float* x     = (const float*)d_in[0];
    const float* W_enc = (const float*)d_in[1];
    const float* b_enc = (const float*)d_in[2];
    const float* W_dec = (const float*)d_in[3];
    const float* b_dec = (const float*)d_in[4];
    // d_in[5] is k; fixed at 128 per the reference constants.

    float* recon = (float*)d_out;                          // [8192 x 2048]
    float* enc   = recon + (size_t)N_TOK * D_MODEL;        // [8192 x 16384]

    float*          WdT  = (float*)d_ws;                         // 16384x2048 f32
    float*          beff = WdT + (size_t)D_SAE * D_MODEL;        // 16384
    float*          topv = beff + D_SAE;                         // 8192x128
    int*            topi = (int*)(topv + (size_t)N_TOK * TOPK);  // 8192x128
    unsigned short* xb   = (unsigned short*)(topi + (size_t)N_TOK * TOPK); // bf16 x
    unsigned short* wb   = xb + (size_t)N_TOK * D_MODEL;         // bf16 W_enc

    k_convert<<<(N_TOK * D_MODEL) / (8 * 256), 256, 0, stream>>>(x, xb);
    k_convert<<<(D_SAE * D_MODEL) / (8 * 256), 256, 0, stream>>>(W_enc, wb);
    k_beff<<<D_SAE / 8, 256, 0, stream>>>(W_enc, b_enc, b_dec, beff);
    k_transpose<<<dim3(D_SAE / 32, D_MODEL / 32), dim3(32, 8), 0, stream>>>(W_dec, WdT);
    k_encoder<<<dim3(D_SAE / BN, N_TOK / BM), 256, 0, stream>>>(xb, wb, beff, enc);
    const size_t smem = (size_t)D_SAE * 4 + 2048 * 4 + TOPK * 8;
    k_topk<<<N_TOK, 256, smem, stream>>>(enc, topv, topi);
    k_decode<<<N_TOK, 256, 0, stream>>>(WdT, b_dec, topv, topi, recon);
}
